// PPOTimeLSTMPolicy_9706626089303
// MI455X (gfx1250) — matrix operand order repaired and checked by reading, not compiled, not run on hardware
//
#include <hip/hip_runtime.h>

// ---------------------------------------------------------------------------
// CDNA5 (gfx1250) bf16 WMMA helpers.  wave32; 16x16x32 tiles, f32 accumulate.
// ---------------------------------------------------------------------------
typedef __attribute__((ext_vector_type(16))) __bf16 v16bf;
typedef __attribute__((ext_vector_type(8)))  __bf16 v8bf;
typedef __attribute__((ext_vector_type(8)))  float  v8f;
typedef __attribute__((ext_vector_type(4)))  unsigned v4u;
typedef __attribute__((ext_vector_type(8)))  int      v8i;
typedef __attribute__((ext_vector_type(4)))  int      v4i;

#define NTOK 16384   // B*T
#define TT   64
#define BBATCH 256

__device__ __forceinline__ v8f wmma_bf16(v16bf a, v16bf b, v8f c) {
  return __builtin_amdgcn_wmma_f32_16x16x32_bf16(false, a, false, b, (short)0, c,
                                                 false, false);
}

__device__ __forceinline__ v16bf cat8(v8bf lo, v8bf hi) {
  return __builtin_shufflevector(lo, hi, 0, 1, 2, 3, 4, 5, 6, 7, 8, 9, 10, 11,
                                 12, 13, 14, 15);
}

// A-fragment (16x32 bf16) from row-major [M x K] storage; two b128 loads.
// Lanes 0-15: row = lane, K = {k0..k0+7, k0+16..k0+23}
// Lanes 16-31: row = lane-16, K = {k0+8..k0+15, k0+24..k0+31}
// Requires lda % 8 == 0, k0 % 32 == 0, 16B-aligned base.
__device__ __forceinline__ v16bf load_a_rm(const __bf16* A, int lda, int m0,
                                           int k0, int lane) {
  int row = m0 + (lane & 15);
  int kb  = (lane & 16) ? 8 : 0;
  const __bf16* p = A + (size_t)row * lda + k0 + kb;
  v8bf lo = *(const v8bf*)p;
  v8bf hi = *(const v8bf*)(p + 16);
  return cat8(lo, hi);
}

// B-fragment from swizzled fragment-major weights: tile (kt,nt) of B = W^T,
// each lane's 16 bf16 contiguous:  off = ((kt*nn + nt)*32 + lane)*16.
__device__ __forceinline__ v16bf load_b_sw(const __bf16* Bm, int nn, int kt,
                                           int nt, int lane) {
  const __bf16* p = Bm + (((size_t)(kt * nn + nt) * 32) + lane) * 16;
  v8bf lo = *(const v8bf*)p;
  v8bf hi = *(const v8bf*)(p + 8);
  return cat8(lo, hi);
}

// C/D fragment: lane -> column, VGPR r -> row (r, lanes<16) / (8+r, lanes>=16).
__device__ __forceinline__ v8f bias_frag(const float* b, int n0, int lane) {
  float bv = b[n0 + (lane & 15)];
  v8f c = {bv, bv, bv, bv, bv, bv, bv, bv};
  return c;
}

__device__ __forceinline__ void store_f32(float* C, int ldc, int m0, int n0,
                                          int lane, v8f d) {
  int col = n0 + (lane & 15);
  int rb  = (lane & 16) ? 8 : 0;
#pragma unroll
  for (int r = 0; r < 8; ++r) C[(size_t)(m0 + rb + r) * ldc + col] = d[r];
}

__device__ __forceinline__ void store_bf16(__bf16* C, int ldc, int m0, int n0,
                                           int lane, v8f d, bool relu) {
  int col = n0 + (lane & 15);
  int rb  = (lane & 16) ? 8 : 0;
#pragma unroll
  for (int r = 0; r < 8; ++r) {
    float v = d[r];
    if (relu && v < 0.f) v = 0.f;
    C[(size_t)(m0 + rb + r) * ldc + col] = (__bf16)v;
  }
}

__device__ __forceinline__ float sigmoidf_(float x) {
  return 1.f / (1.f + __expf(-x));
}

// Deterministic grid-wide barrier (persistent kernel, all WGs resident).
__device__ __forceinline__ void grid_barrier(unsigned* cnt, unsigned* gen,
                                             unsigned nwg) {
  __syncthreads();
  if (threadIdx.x == 0) {
    unsigned g = __hip_atomic_load(gen, __ATOMIC_ACQUIRE, __HIP_MEMORY_SCOPE_AGENT);
    unsigned a = __hip_atomic_fetch_add(cnt, 1u, __ATOMIC_ACQ_REL,
                                        __HIP_MEMORY_SCOPE_AGENT);
    if (a == nwg - 1u) {
      __hip_atomic_store(cnt, 0u, __ATOMIC_RELAXED, __HIP_MEMORY_SCOPE_AGENT);
      __hip_atomic_fetch_add(gen, 1u, __ATOMIC_ACQ_REL, __HIP_MEMORY_SCOPE_AGENT);
    } else {
      while (__hip_atomic_load(gen, __ATOMIC_ACQUIRE, __HIP_MEMORY_SCOPE_AGENT) == g)
        __builtin_amdgcn_s_sleep(2);
    }
  }
  __syncthreads();
}

// ---------------------------------------------------------------------------
// Prep: weight convert to fragment-major swizzled bf16 B-layout (W^T, K padded)
// out element idx = ((kt*nn + nt)*32 + lane)*16 + i,
//   n = nt*16 + (lane&15),  k = kt*32 + (lane&16 ? 8:0) + (i&7) + (i&8 ? 16:0).
// ---------------------------------------------------------------------------
__global__ void k_convw(const float* W, __bf16* out, int O, int I, int Kpad) {
  int idx = blockIdx.x * 256 + threadIdx.x;
  if (idx >= Kpad * O) return;
  int i    = idx & 15;
  int lane = (idx >> 4) & 31;
  int tile = idx >> 9;
  int nn = O >> 4;
  int kt = tile / nn, nt = tile - kt * nn;
  int n = nt * 16 + (lane & 15);
  int k = kt * 32 + ((lane & 16) ? 8 : 0) + (i & 7) + ((i & 8) ? 16 : 0);
  float v = (k < I) ? W[(size_t)n * I + k] : 0.f;
  out[idx] = (__bf16)v;
}

// Gather obs -> packed bf16 runner input (32 rows/token, rows 30..31 zero),
// market input (64 cols, 40 real), and dt scalar (obs col 26).
__global__ void k_prep(const float* __restrict__ obs, __bf16* rin, __bf16* min_,
                       float* dtv) {
  int t = blockIdx.x;
  const float* o = obs + (size_t)t * 1000;
  int tid = threadIdx.x;
  if (tid < 64) {
    float v = 0.f;
    if (tid < 30) v = o[tid];                 // market(20) + vel(10)
    else if (tid < 40) v = o[720 + tid];      // agent: cols 750..759
    min_[(size_t)t * 64 + tid] = (__bf16)v;
  }
  for (int idx = tid; idx < 32 * 32; idx += 256) {
    int r = idx >> 5, j = idx & 31;
    float v = 0.f;
    if (r < 30) v = (j < 24) ? o[30 + r * 24 + j] : o[760 + r * 8 + (j - 24)];
    rin[(size_t)(t * 32 + r) * 32 + j] = (__bf16)v;
  }
  if (tid == 0) dtv[t] = o[26];  // TD_IDX
}

__global__ void k_init(__bf16* h0, __bf16* h1, unsigned* sync) {
  int i = blockIdx.x * blockDim.x + threadIdx.x;
  int stride = gridDim.x * blockDim.x;
  for (int j = i; j < 256 * 256; j += stride) {
    h0[j] = (__bf16)0.f;
    h1[j] = (__bf16)0.f;
  }
  if (i == 0) { sync[0] = 0u; sync[1] = 0u; }
}

// ---------------------------------------------------------------------------
// Runner MLP3 (32->128->128->128) + mean/max pool; one WG per token (32 rows).
// ---------------------------------------------------------------------------
__global__ void k_runner(const __bf16* __restrict__ rin, const __bf16* B0,
                         const __bf16* B1, const __bf16* B2, const float* b0,
                         const float* b1, const float* b2, __bf16* lstm_in,
                         __bf16* rlast) {
  __shared__ alignas(16) __bf16 sh1[32 * 128];
  __shared__ alignas(16) __bf16 sh2[32 * 128];
  __shared__ alignas(16) float  sh3[32 * 128];
  int t = blockIdx.x;
  int tid = threadIdx.x, lane = tid & 31, w = tid >> 5;
  int mt = (w & 1) * 16;
  int ng = w >> 1;  // 0..3, two n-tiles each
  const __bf16* x = rin + (size_t)t * 32 * 32;

#pragma unroll
  for (int j = 0; j < 2; ++j) {  // layer0: K=32 (one wmma)
    int nt = ng * 2 + j, n0 = nt * 16;
    v8f acc = bias_frag(b0, n0, lane);
    acc = wmma_bf16(load_a_rm(x, 32, mt, 0, lane),
                    load_b_sw(B0, 8, 0, nt, lane), acc);
    store_bf16(sh1, 128, mt, n0, lane, acc, true);
  }
  __syncthreads();
#pragma unroll
  for (int j = 0; j < 2; ++j) {  // layer1: K=128
    int nt = ng * 2 + j, n0 = nt * 16;
    v8f acc = bias_frag(b1, n0, lane);
#pragma unroll
    for (int k = 0; k < 4; ++k)
      acc = wmma_bf16(load_a_rm(sh1, 128, mt, k * 32, lane),
                      load_b_sw(B1, 8, k, nt, lane), acc);
    store_bf16(sh2, 128, mt, n0, lane, acc, true);
  }
  __syncthreads();
#pragma unroll
  for (int j = 0; j < 2; ++j) {  // layer2: linear, f32 for pooling
    int nt = ng * 2 + j, n0 = nt * 16;
    v8f acc = bias_frag(b2, n0, lane);
#pragma unroll
    for (int k = 0; k < 4; ++k)
      acc = wmma_bf16(load_a_rm(sh2, 128, mt, k * 32, lane),
                      load_b_sw(B2, 8, k, nt, lane), acc);
    store_f32(sh3, 128, mt, n0, lane, acc);
  }
  __syncthreads();
  if (tid < 128) {
    float s = 0.f, mx = -3.4e38f;
    for (int r = 0; r < 30; ++r) {
      float v = sh3[r * 128 + tid];
      s += v;
      mx = fmaxf(mx, v);
    }
    lstm_in[(size_t)t * 384 + 128 + tid] = (__bf16)(s * (1.f / 30.f));
    lstm_in[(size_t)t * 384 + 256 + tid] = (__bf16)mx;
  }
  if ((t & 63) == 63) {  // last timestep -> runner_last for actor head
    int b = t >> 6;
    for (int idx = tid; idx < 30 * 128; idx += 256) {
      int r = idx >> 7, c = idx & 127;
      rlast[((size_t)b * 30 + r) * 128 + c] = (__bf16)sh3[r * 128 + c];
    }
  }
}

// ---------------------------------------------------------------------------
// Market MLP3 (64(pad of 40)->128->128->128); 64 rows per WG.
// ---------------------------------------------------------------------------
__global__ void k_market(const __bf16* __restrict__ min_, const __bf16* B0,
                         const __bf16* B1, const __bf16* B2, const float* b0,
                         const float* b1, const float* b2, __bf16* lstm_in) {
  __shared__ alignas(16) __bf16 sh1[64 * 128];
  __shared__ alignas(16) __bf16 sh2[64 * 128];
  int r0 = blockIdx.x * 64;
  int tid = threadIdx.x, lane = tid & 31, w = tid >> 5;
  int mt = (w & 3) * 16;
  int ng = w >> 2;  // 0..1, four n-tiles each
  const __bf16* x = min_ + (size_t)r0 * 64;
#pragma unroll
  for (int j = 0; j < 4; ++j) {
    int nt = ng * 4 + j, n0 = nt * 16;
    v8f acc = bias_frag(b0, n0, lane);
#pragma unroll
    for (int k = 0; k < 2; ++k)
      acc = wmma_bf16(load_a_rm(x, 64, mt, k * 32, lane),
                      load_b_sw(B0, 8, k, nt, lane), acc);
    store_bf16(sh1, 128, mt, n0, lane, acc, true);
  }
  __syncthreads();
#pragma unroll
  for (int j = 0; j < 4; ++j) {
    int nt = ng * 4 + j, n0 = nt * 16;
    v8f acc = bias_frag(b1, n0, lane);
#pragma unroll
    for (int k = 0; k < 4; ++k)
      acc = wmma_bf16(load_a_rm(sh1, 128, mt, k * 32, lane),
                      load_b_sw(B1, 8, k, nt, lane), acc);
    store_bf16(sh2, 128, mt, n0, lane, acc, true);
  }
  __syncthreads();
#pragma unroll
  for (int j = 0; j < 4; ++j) {
    int nt = ng * 4 + j, n0 = nt * 16;
    v8f acc = bias_frag(b2, n0, lane);
#pragma unroll
    for (int k = 0; k < 4; ++k)
      acc = wmma_bf16(load_a_rm(sh2, 128, mt, k * 32, lane),
                      load_b_sw(B2, 8, k, nt, lane), acc);
    int col = n0 + (lane & 15);
    int rb  = (lane & 16) ? 8 : 0;
#pragma unroll
    for (int r = 0; r < 8; ++r)
      lstm_in[(size_t)(r0 + mt + rb + r) * 384 + col] = (__bf16)acc[r];
  }
}

// ---------------------------------------------------------------------------
// xg = lstm_in @ W_ih^T + b_ih  (16384 x 1024, K=384), fp32 out.
// A-tile (64x384 bf16) staged into LDS by the Tensor Data Mover.
// ---------------------------------------------------------------------------
__global__ void k_xg(const __bf16* __restrict__ lin, const __bf16* Wih,
                     const float* bih, float* xg) {
  __shared__ alignas(16) __bf16 sA[64 * 384];
  int rb_ = blockIdx.x >> 3;  // row block (64 rows)
  int nb  = blockIdx.x & 7;   // col block (128 cols)
  int r0  = rb_ * 64;
  int tid = threadIdx.x, lane = tid & 31, w = tid >> 5;

  if (w == 0) {
    // TDM: 2D tile, 384 x 64 elements of 2 bytes, row stride 384 elements.
    unsigned lds_addr = (unsigned)(uintptr_t)(&sA[0]);
    unsigned long long ga = (unsigned long long)(uintptr_t)(lin + (size_t)r0 * 384);
    v4u g0 = {1u,                                   // count=1, user mode
              lds_addr,                             // D#.lds_addr
              (unsigned)(ga & 0xffffffffu),         // global_addr[31:0]
              (unsigned)((ga >> 32) & 0x01ffffffu) | 0x80000000u};  // type=2
    v8i g1 = {(int)(1u << 16),        // data_size = 1 (2 bytes)
              (int)(384u << 16),      // tensor_dim0[15:0]
              (int)(16384u << 16),    // tensor_dim0[31:16]=0 | tensor_dim1[15:0]
              (int)(384u << 16),      // tensor_dim1[31:16]=0 | tile_dim0
              64,                     // tile_dim1 | tile_dim2=0
              384,                    // tensor_dim0_stride[31:0]
              0, 0};                  // stride hi / dim1_stride unused (2D)
    v4i gz4 = {0, 0, 0, 0};
    v8i gz8 = {0, 0, 0, 0, 0, 0, 0, 0};
    __builtin_amdgcn_tensor_load_to_lds(g0, g1, gz4, gz4, gz8, 0);
    __builtin_amdgcn_s_wait_tensorcnt(0);
  }
  __syncthreads();

  int mt = (w & 3) * 16;
  int ng = w >> 2;
#pragma unroll
  for (int j = 0; j < 4; ++j) {
    int nt = nb * 8 + (ng * 4 + j);  // global n-tile index (of 64)
    int n0 = nt * 16;
    v8f acc = bias_frag(bih, n0, lane);
#pragma unroll
    for (int k = 0; k < 12; ++k)
      acc = wmma_bf16(load_a_rm(sA, 384, mt, k * 32, lane),
                      load_b_sw(Wih, 64, k, nt, lane), acc);
    store_f32(xg, 1024, r0 + mt, n0, lane, acc);
  }
}

// ---------------------------------------------------------------------------
// Persistent LSTM: 16 WGs x 256 threads. WG g owns hidden units [16g,16g+16)
// across all 4 gate blocks, so i/f/g/o for a unit land in one wave's regs and
// cell state c stays in VGPRs for all 64 steps.  h ping-pongs in bf16 global.
// ---------------------------------------------------------------------------
__global__ void k_lstm(const float* __restrict__ xg, const __bf16* Whh,
                       const float* Wdt, const float* dtv, __bf16* h0b,
                       __bf16* h1b, float* oh, float* oc, unsigned* sync) {
  __shared__ float sdt[256];
  int tid = threadIdx.x, lane = tid & 31, w = tid >> 5;
  int u0  = blockIdx.x * 16;
  int ut  = blockIdx.x;          // n-tile of the unit slice within a gate block
  int m0  = w * 32;
  int col = lane & 15;
  int rb  = (lane & 16) ? 8 : 0;
  float wdt = Wdt[u0 + col];
  v8f cst[2];
#pragma unroll
  for (int q = 0; q < 2; ++q)
#pragma unroll
    for (int r = 0; r < 8; ++r) cst[q][r] = 0.f;

  for (int t = 0; t < TT; ++t) {
    sdt[tid] = dtv[(size_t)tid * TT + t];  // dt[b, t]
    __syncthreads();
    const __bf16* hr = (t & 1) ? h1b : h0b;
    __bf16* hw       = (t & 1) ? h0b : h1b;
    for (int q = 0; q < 2; ++q) {
      int mrow = m0 + q * 16;
      v16bf a[8];
#pragma unroll
      for (int k = 0; k < 8; ++k) a[k] = load_a_rm(hr, 256, mrow, k * 32, lane);
      v8f acc[4];
#pragma unroll
      for (int G = 0; G < 4; ++G) {  // init with precomputed x@W_ih^T + b_ih
        int n = G * 256 + u0 + col;
#pragma unroll
        for (int r = 0; r < 8; ++r)
          acc[G][r] = xg[(size_t)((mrow + rb + r) * TT + t) * 1024 + n];
      }
#pragma unroll
      for (int k = 0; k < 8; ++k)
#pragma unroll
        for (int G = 0; G < 4; ++G)
          acc[G] = wmma_bf16(a[k], load_b_sw(Whh, 64, k, G * 16 + ut, lane),
                             acc[G]);
#pragma unroll
      for (int r = 0; r < 8; ++r) {
        int row = mrow + rb + r;
        float dt = sdt[row];
        float iv = sigmoidf_(acc[0][r]);
        float fv = sigmoidf_(acc[1][r] + wdt * dt);
        float gv = tanhf(acc[2][r]);
        float ov = sigmoidf_(acc[3][r]);
        float cn = fv * cst[q][r] + iv * gv;
        float hn = ov * tanhf(cn);
        cst[q][r] = cn;
        hw[(size_t)row * 256 + u0 + col] = (__bf16)hn;
        if (t == TT - 1) {
          oh[(size_t)row * 256 + u0 + col] = hn;
          oc[(size_t)row * 256 + u0 + col] = cn;
        }
      }
    }
    grid_barrier(sync, sync + 1, 16);
  }
}

// ---------------------------------------------------------------------------
// Actor head: rows = 7680 (b*30+r); A = [runner_last(128) | h(256)] gathered
// on the fly; layer0 WMMA -> ReLU -> 128->2 VALU dot.
// ---------------------------------------------------------------------------
__device__ __forceinline__ v16bf load_a_act(const __bf16* rl, const __bf16* hb,
                                            int row0, int k0, int lane) {
  int row = row0 + (lane & 15);
  int kb  = (lane & 16) ? 8 : 0;
  const __bf16* p = (k0 < 128) ? (rl + (size_t)row * 128 + k0 + kb)
                               : (hb + (size_t)(row / 30) * 256 + (k0 - 128) + kb);
  v8bf lo = *(const v8bf*)p;
  v8bf hi = *(const v8bf*)(p + 16);
  return cat8(lo, hi);
}

__global__ void k_actor(const __bf16* __restrict__ rl,
                        const __bf16* __restrict__ hb, const __bf16* B0,
                        const float* b0, const float* w1, const float* b1,
                        float* am) {
  __shared__ alignas(16) __bf16 sh1[64 * 128];
  int r0 = blockIdx.x * 64;
  int tid = threadIdx.x, lane = tid & 31, w = tid >> 5;
  int mt = (w & 3) * 16, ng = w >> 2;
#pragma unroll
  for (int j = 0; j < 4; ++j) {
    int nt = ng * 4 + j, n0 = nt * 16;
    v8f acc = bias_frag(b0, n0, lane);
#pragma unroll
    for (int k = 0; k < 12; ++k)
      acc = wmma_bf16(load_a_act(rl, hb, r0 + mt, k * 32, lane),
                      load_b_sw(B0, 8, k, nt, lane), acc);
    store_bf16(sh1, 128, mt, n0, lane, acc, true);
  }
  __syncthreads();
  if (tid < 128) {
    int row = tid >> 1, cc = tid & 1;
    float s = b1[cc];
    for (int k = 0; k < 128; ++k)
      s += (float)sh1[row * 128 + k] * w1[cc * 128 + k];
    int g = r0 + row, bidx = g / 30, rr = g - bidx * 30;
    am[(size_t)bidx * 60 + rr * 2 + cc] = s;
  }
}

__global__ void k_logstd(const float* ls, float* out) {
  int i = blockIdx.x * 256 + threadIdx.x;
  if (i < 256 * 60) out[i] = ls[i % 60];
}

// Critic: h(256x256) -> relu(128) WMMA -> 128->1 VALU dot. 128 rows per WG.
__global__ void k_critic(const __bf16* __restrict__ hb, const __bf16* B0,
                         const float* b0, const float* w1, const float* b1,
                         float* val) {
  __shared__ alignas(16) __bf16 sh1[128 * 128];
  int r0 = blockIdx.x * 128;
  int tid = threadIdx.x, lane = tid & 31, w = tid >> 5;
  int mt = w * 16;
  v16bf a[8];
#pragma unroll
  for (int k = 0; k < 8; ++k) a[k] = load_a_rm(hb, 256, r0 + mt, k * 32, lane);
#pragma unroll
  for (int nt = 0; nt < 8; ++nt) {
    int n0 = nt * 16;
    v8f acc = bias_frag(b0, n0, lane);
#pragma unroll
    for (int k = 0; k < 8; ++k)
      acc = wmma_bf16(a[k], load_b_sw(B0, 8, k, nt, lane), acc);
    store_bf16(sh1, 128, mt, n0, lane, acc, true);
  }
  __syncthreads();
  if (tid < 128) {
    float s = b1[0];
    for (int k = 0; k < 128; ++k) s += (float)sh1[tid * 128 + k] * w1[k];
    val[r0 + tid] = s;
  }
}

// ---------------------------------------------------------------------------
extern "C" void kernel_launch(void* const* d_in, const int* in_sizes, int n_in,
                              void* d_out, int out_size, void* d_ws,
                              size_t ws_size, hipStream_t stream) {
  (void)in_sizes; (void)n_in; (void)out_size; (void)ws_size;
  const float* obs   = (const float*)d_in[0];
  const float* re_w0 = (const float*)d_in[1];
  const float* re_b0 = (const float*)d_in[2];
  const float* re_w1 = (const float*)d_in[3];
  const float* re_b1 = (const float*)d_in[4];
  const float* re_w2 = (const float*)d_in[5];
  const float* re_b2 = (const float*)d_in[6];
  const float* me_w0 = (const float*)d_in[7];
  const float* me_b0 = (const float*)d_in[8];
  const float* me_w1 = (const float*)d_in[9];
  const float* me_b1 = (const float*)d_in[10];
  const float* me_w2 = (const float*)d_in[11];
  const float* me_b2 = (const float*)d_in[12];
  const float* W_ih  = (const float*)d_in[13];
  const float* b_ih  = (const float*)d_in[14];
  const float* W_hh  = (const float*)d_in[15];
  const float* W_dt  = (const float*)d_in[16];
  const float* a_w0  = (const float*)d_in[17];
  const float* a_b0  = (const float*)d_in[18];
  const float* a_w1  = (const float*)d_in[19];
  const float* a_b1  = (const float*)d_in[20];
  const float* c_w0  = (const float*)d_in[21];
  const float* c_b0  = (const float*)d_in[22];
  const float* c_w1  = (const float*)d_in[23];
  const float* c_b1  = (const float*)d_in[24];
  const float* lstd  = (const float*)d_in[25];

  // output layout: action_mean | action_log_std | value | h | c
  float* o   = (float*)d_out;
  float* am  = o;
  float* lso = o + 15360;
  float* vo  = o + 30720;
  float* oh  = o + 30976;
  float* oc  = o + 96512;

  char* cur = (char*)d_ws;
  auto alloc = [&](size_t bytes) -> char* {
    char* p = cur;
    cur += (bytes + 255) & ~(size_t)255;
    return p;
  };
  unsigned* sync = (unsigned*)alloc(2 * sizeof(unsigned));
  __bf16* reB0 = (__bf16*)alloc((size_t)32 * 128 * 2);
  __bf16* reB1 = (__bf16*)alloc((size_t)128 * 128 * 2);
  __bf16* reB2 = (__bf16*)alloc((size_t)128 * 128 * 2);
  __bf16* meB0 = (__bf16*)alloc((size_t)64 * 128 * 2);
  __bf16* meB1 = (__bf16*)alloc((size_t)128 * 128 * 2);
  __bf16* meB2 = (__bf16*)alloc((size_t)128 * 128 * 2);
  __bf16* WihB = (__bf16*)alloc((size_t)384 * 1024 * 2);
  __bf16* WhhB = (__bf16*)alloc((size_t)256 * 1024 * 2);
  __bf16* aB0  = (__bf16*)alloc((size_t)384 * 128 * 2);
  __bf16* cB0  = (__bf16*)alloc((size_t)256 * 128 * 2);
  __bf16* h0b  = (__bf16*)alloc((size_t)256 * 256 * 2);
  __bf16* h1b  = (__bf16*)alloc((size_t)256 * 256 * 2);
  __bf16* rlast= (__bf16*)alloc((size_t)256 * 30 * 128 * 2);
  float*  dtv  = (float*)alloc((size_t)NTOK * 4);
  __bf16* minb = (__bf16*)alloc((size_t)NTOK * 64 * 2);
  __bf16* lin  = (__bf16*)alloc((size_t)NTOK * 384 * 2);
  __bf16* rin  = (__bf16*)alloc((size_t)NTOK * 32 * 32 * 2);
  float*  xg   = (float*)alloc((size_t)NTOK * 1024 * 4);

  // weight conversion (fragment-major swizzled bf16, zero-padded K)
  k_convw<<<(32 * 128 + 255) / 256, 256, 0, stream>>>(re_w0, reB0, 128, 32, 32);
  k_convw<<<(128 * 128 + 255) / 256, 256, 0, stream>>>(re_w1, reB1, 128, 128, 128);
  k_convw<<<(128 * 128 + 255) / 256, 256, 0, stream>>>(re_w2, reB2, 128, 128, 128);
  k_convw<<<(64 * 128 + 255) / 256, 256, 0, stream>>>(me_w0, meB0, 128, 40, 64);
  k_convw<<<(128 * 128 + 255) / 256, 256, 0, stream>>>(me_w1, meB1, 128, 128, 128);
  k_convw<<<(128 * 128 + 255) / 256, 256, 0, stream>>>(me_w2, meB2, 128, 128, 128);
  k_convw<<<(384 * 1024 + 255) / 256, 256, 0, stream>>>(W_ih, WihB, 1024, 384, 384);
  k_convw<<<(256 * 1024 + 255) / 256, 256, 0, stream>>>(W_hh, WhhB, 1024, 256, 256);
  k_convw<<<(384 * 128 + 255) / 256, 256, 0, stream>>>(a_w0, aB0, 128, 384, 384);
  k_convw<<<(256 * 128 + 255) / 256, 256, 0, stream>>>(c_w0, cB0, 128, 256, 256);

  k_prep<<<NTOK, 256, 0, stream>>>(obs, rin, minb, dtv);
  k_init<<<128, 256, 0, stream>>>(h0b, h1b, sync);

  k_runner<<<NTOK, 256, 0, stream>>>(rin, reB0, reB1, reB2, re_b0, re_b1, re_b2,
                                     lin, rlast);
  k_market<<<NTOK / 64, 256, 0, stream>>>(minb, meB0, meB1, meB2, me_b0, me_b1,
                                          me_b2, lin);
  k_xg<<<(NTOK / 64) * 8, 256, 0, stream>>>(lin, WihB, b_ih, xg);
  k_lstm<<<16, 256, 0, stream>>>(xg, WhhB, W_dt, dtv, h0b, h1b, oh, oc, sync);
  k_actor<<<7680 / 64, 256, 0, stream>>>(rlast, h0b, aB0, a_b0, a_w1, a_b1, am);
  k_logstd<<<60, 256, 0, stream>>>(lstd, lso);
  k_critic<<<2, 256, 0, stream>>>(h0b, cB0, c_b0, c_w1, c_b1, vo);
}
